// BallCrop_55310588838566
// MI455X (gfx1250) — matrix operand order repaired and checked by reading, not compiled, not run on hardware
//
#include <hip/hip_runtime.h>
#include <hip/hip_bf16.h>
#include <math.h>

// Problem constants (match reference)
#define BB 4
#define NN 16384
#define MM 1024
#define CC 256
#define SS 64          // nsample
#define KP 288         // 3 + 256 = 259 padded to 9*32 (GEMM K extent)
#define KPAD 296       // LDS row stride for Xs: 148 DW % 64 = 20 -> conflict-free starts
#define OC 256
#define OCPAD 264      // LDS row stride for Y1s: 132 DW % 64 = 4 -> conflict-free starts
#define R2 (0.08f * 0.08f)
#define EPSV 1e-5f

typedef _Float16 v16h __attribute__((ext_vector_type(16)));
typedef _Float16 h8   __attribute__((ext_vector_type(8)));
typedef float    v8f  __attribute__((ext_vector_type(8)));

union AFrag { v16h v; h8 h[2]; };

// ---------------------------------------------------------------------------
// Kernel 1: fold BatchNorm into weights; f32 -> f16, pad K to 288 for layer 1.
// grid = 256 (one block per output channel), block = 256
// ---------------------------------------------------------------------------
__global__ void fold_weights_kernel(
    const float* __restrict__ W1, const float* __restrict__ b1,
    const float* __restrict__ g1, const float* __restrict__ be1,
    const float* __restrict__ m1, const float* __restrict__ v1,
    const float* __restrict__ W2, const float* __restrict__ b2,
    const float* __restrict__ g2, const float* __restrict__ be2,
    const float* __restrict__ m2, const float* __restrict__ v2,
    _Float16* __restrict__ W1h, _Float16* __restrict__ W2h,
    float* __restrict__ bias1, float* __restrict__ bias2)
{
    const int o = blockIdx.x;
    const int t = threadIdx.x;
    const float s1 = g1[o] * rsqrtf(v1[o] + EPSV);
    const float s2 = g2[o] * rsqrtf(v2[o] + EPSV);
    if (t == 0) {
        bias1[o] = (b1[o] - m1[o]) * s1 + be1[o];
        bias2[o] = (b2[o] - m2[o]) * s2 + be2[o];
    }
    for (int k = t; k < KP; k += 256) {
        float w = (k < 259) ? W1[o * 259 + k] : 0.0f;
        W1h[(size_t)o * KP + k] = (_Float16)(w * s1);
    }
    for (int k = t; k < 256; k += 256) {
        W2h[(size_t)o * 256 + k] = (_Float16)(W2[o * 256 + k] * s2);
    }
}

// ---------------------------------------------------------------------------
// Kernel 2: ball query. One wave32 per query point; ordered index append via
// ballot + prefix popcount (preserves "first nsample in point order" semantics,
// padding with the first found index, 0 if none).
// grid = B*M/8 blocks of 256 threads (8 waves)
// ---------------------------------------------------------------------------
__global__ void ball_query_kernel(const float* __restrict__ xyz,
                                  const float* __restrict__ new_xyz,
                                  int* __restrict__ idxbuf)
{
    const int gwave = (blockIdx.x * blockDim.x + threadIdx.x) >> 5;
    const int lane  = threadIdx.x & 31;
    if (gwave >= BB * MM) return;
    const int b = gwave >> 10;          // / MM

    const float qx = new_xyz[gwave * 3 + 0];
    const float qy = new_xyz[gwave * 3 + 1];
    const float qz = new_xyz[gwave * 3 + 2];

    const float* __restrict__ p = xyz + (size_t)b * NN * 3;
    int* __restrict__ out = idxbuf + (size_t)gwave * SS;

    int cnt = 0;
    int first = 0;
    for (int n0 = 0; n0 < NN && cnt < SS; n0 += 32) {
        const int n = n0 + lane;
        const float dx = p[n * 3 + 0] - qx;
        const float dy = p[n * 3 + 1] - qy;
        const float dz = p[n * 3 + 2] - qz;
        const bool pred = (dx * dx + dy * dy + dz * dz) < R2;
        const unsigned mlo = (unsigned)__ballot(pred);   // wave32: low 32 bits
        if (cnt == 0 && mlo != 0u) first = n0 + __ffs(mlo) - 1;
        const int before = __popc(mlo & ((1u << lane) - 1u));
        const int slot = cnt + before;
        if (pred && slot < SS) out[slot] = n;
        cnt += __popc(mlo);
    }
    // pad remaining slots with first index (0 if none found)
    for (int s = cnt + lane; s < SS; s += 32) out[s] = first;
}

// ---------------------------------------------------------------------------
// Kernel 3: fused gather + (GEMM1 + bias + ReLU) + (GEMM2 + bias + ReLU) + max.
// One workgroup (8 wave32) per (b,m) group of 64 samples.
// Wave w owns row-tile rt = w>>1 (16 rows) and col-tiles (w&1)*8 .. +8.
// WMMA layouts used here (A/B in contiguous K order, not the ISA table order):
//   A 16x32 f16 : lanes 0-15 row=lane, halves = K[0..7],K[8..15];
//                 lanes 16-31 row=lane-16, halves = K[16..23],K[24..31]
//   B 32x16 f16 : lanes 0-15 col=lane, halves = K[0..15];
//                 lanes 16-31 col=lane-16, halves = K[16..31]
//   C/D f32     : VGPR j, lanes 0-15 -> (M=j, N=lane); lanes 16-31 -> (M=j+8)
// Inner loop issues all 8 B-fragment loads (16x b128) before the 8-WMMA chain
// so vector loads overlap the matrix pipe instead of 2-deep wait pairs.
// ---------------------------------------------------------------------------
__global__ __launch_bounds__(256)
void fused_mlp_kernel(const float* __restrict__ xyz,
                      const float* __restrict__ new_xyz,
                      const float* __restrict__ feats,      // [B,C,N]
                      const int*   __restrict__ idxbuf,     // [B*M,64]
                      const _Float16* __restrict__ W1h,     // [256][288]
                      const float* __restrict__ bias1,
                      const _Float16* __restrict__ W2h,     // [256][256]
                      const float* __restrict__ bias2,
                      float* __restrict__ out)              // [B,256,M]
{
    __shared__ alignas(32) _Float16 Xs[SS][KPAD];    // 37,888 B (padded stride)
    __shared__ alignas(32) _Float16 Y1s[SS][OCPAD];  // 33,792 B (padded stride)
    __shared__ float pm[4][OC];                      //  4,096 B
    __shared__ int sidx[SS];

    const int t  = threadIdx.x;
    const int bm = blockIdx.x;            // 0 .. B*M-1
    const int b  = bm >> 10;
    const int m  = bm & (MM - 1);

    if (t < SS) sidx[t] = idxbuf[(size_t)bm * SS + t];

    // warm WGP$ with the folded weights while we gather (global_prefetch_b8)
    {
        const char* w1p = (const char*)W1h;
        const char* w2p = (const char*)W2h;
        for (int off = t * 64; off < 256 * KP * 2; off += 256 * 64)
            __builtin_prefetch(w1p + off, 0, 0);
        for (int off = t * 64; off < 256 * OC * 2; off += 256 * 64)
            __builtin_prefetch(w2p + off, 0, 0);
    }
    __syncthreads();

    const float qx = new_xyz[bm * 3 + 0];
    const float qy = new_xyz[bm * 3 + 1];
    const float qz = new_xyz[bm * 3 + 2];

    // ---- gather: relative xyz + K-padding (threads 0..63), features (all) ----
    if (t < SS) {
        const int ns = sidx[t];
        const float* p = xyz + ((size_t)b * NN + ns) * 3;
        Xs[t][0] = (_Float16)(p[0] - qx);
        Xs[t][1] = (_Float16)(p[1] - qy);
        Xs[t][2] = (_Float16)(p[2] - qz);
        #pragma unroll
        for (int k = 259; k < KP; ++k) Xs[t][k] = (_Float16)0.0f;
    }
    {
        const int s  = t & 63;            // sample handled by this thread
        const int cg = t >> 6;            // channel group 0..3
        const int ns = sidx[s];
        const float* f = feats + (size_t)b * CC * NN + ns;  // f[c*NN]
        #pragma unroll 4
        for (int i = 0; i < 64; ++i) {
            const int c = cg * 64 + i;
            Xs[s][3 + c] = (_Float16)f[(size_t)c * NN];
        }
    }
    __syncthreads();

    const int wv    = t >> 5;             // wave 0..7
    const int lane  = t & 31;
    const int lhalf = lane >> 4;          // 0 = lo half, 1 = hi half
    const int l16   = lane & 15;
    const int rt    = wv >> 1;            // row tile 0..3
    const int cbase = (wv & 1) * 8;       // first col-tile of this wave

    const int arow  = rt * 16 + l16;
    const int akoff = lhalf * 16;         // A: lo lanes K0..15, hi K16..31
    const int bkoff = lhalf * 16;         // B: lo lanes K0..15, hi K16..31

    v8f acc[8];
    #pragma unroll
    for (int ct = 0; ct < 8; ++ct) { v8f z = {}; acc[ct] = z; }

    // ---------------- layer 1: [64,288] x [288,256] ----------------
    #pragma unroll 1
    for (int k0 = 0; k0 < KP; k0 += 32) {
        AFrag a;
        a.h[0] = *(const h8*)&Xs[arow][k0 + akoff];
        a.h[1] = *(const h8*)&Xs[arow][k0 + akoff + 8];
        v16h bf[8];
        #pragma unroll
        for (int ct = 0; ct < 8; ++ct) {
            const int n = (cbase + ct) * 16 + l16;
            bf[ct] = *(const v16h*)&W1h[(size_t)n * KP + k0 + bkoff];
        }
        #pragma unroll
        for (int ct = 0; ct < 8; ++ct) {
            acc[ct] = __builtin_amdgcn_wmma_f32_16x16x32_f16(
                false, a.v, false, bf[ct], (short)0, acc[ct], false, false);
        }
    }
    // bias + ReLU -> f16 LDS activation tile
    #pragma unroll
    for (int ct = 0; ct < 8; ++ct) {
        const int col = (cbase + ct) * 16 + l16;
        const float bs = bias1[col];
        #pragma unroll
        for (int j = 0; j < 8; ++j) {
            const int row = rt * 16 + j + lhalf * 8;
            float v = acc[ct][j] + bs;
            v = v > 0.0f ? v : 0.0f;
            Y1s[row][col] = (_Float16)v;
        }
    }
    __syncthreads();

    // ---------------- layer 2: [64,256] x [256,256] ----------------
    #pragma unroll
    for (int ct = 0; ct < 8; ++ct) { v8f z = {}; acc[ct] = z; }
    #pragma unroll 1
    for (int k0 = 0; k0 < OC; k0 += 32) {
        AFrag a;
        a.h[0] = *(const h8*)&Y1s[arow][k0 + akoff];
        a.h[1] = *(const h8*)&Y1s[arow][k0 + akoff + 8];
        v16h bf[8];
        #pragma unroll
        for (int ct = 0; ct < 8; ++ct) {
            const int n = (cbase + ct) * 16 + l16;
            bf[ct] = *(const v16h*)&W2h[(size_t)n * OC + k0 + bkoff];
        }
        #pragma unroll
        for (int ct = 0; ct < 8; ++ct) {
            acc[ct] = __builtin_amdgcn_wmma_f32_16x16x32_f16(
                false, a.v, false, bf[ct], (short)0, acc[ct], false, false);
        }
    }
    // bias + ReLU + max over this wave's 16 rows, store per-rowtile partials
    #pragma unroll
    for (int ct = 0; ct < 8; ++ct) {
        const int col = (cbase + ct) * 16 + l16;
        const float bs = bias2[col];
        float mx = 0.0f;                       // ReLU floor
        #pragma unroll
        for (int j = 0; j < 8; ++j) {
            float v = acc[ct][j] + bs;
            v = v > 0.0f ? v : 0.0f;
            mx = fmaxf(mx, v);
        }
        const float other = __shfl_xor(mx, 16, 32);  // merge row halves
        mx = fmaxf(mx, other);
        if (lhalf == 0) pm[rt][col] = mx;
    }
    __syncthreads();

    // final max across the 4 row tiles; out[b, ch, m]
    {
        const float r = fmaxf(fmaxf(pm[0][t], pm[1][t]),
                              fmaxf(pm[2][t], pm[3][t]));
        out[((size_t)b * OC + t) * MM + m] = r;
    }
}

// ---------------------------------------------------------------------------
// Launch
// ---------------------------------------------------------------------------
extern "C" void kernel_launch(void* const* d_in, const int* in_sizes, int n_in,
                              void* d_out, int out_size, void* d_ws, size_t ws_size,
                              hipStream_t stream) {
    (void)in_sizes; (void)n_in; (void)out_size; (void)ws_size;

    const float* xyz   = (const float*)d_in[0];   // [B,N,3]
    const float* nxyz  = (const float*)d_in[1];   // [B,M,3]
    const float* feats = (const float*)d_in[2];   // [B,C,N]
    const float* W1  = (const float*)d_in[3];
    const float* b1  = (const float*)d_in[4];
    const float* g1  = (const float*)d_in[5];
    const float* be1 = (const float*)d_in[6];
    const float* m1  = (const float*)d_in[7];
    const float* v1  = (const float*)d_in[8];
    const float* W2  = (const float*)d_in[9];
    const float* b2  = (const float*)d_in[10];
    const float* g2  = (const float*)d_in[11];
    const float* be2 = (const float*)d_in[12];
    const float* m2  = (const float*)d_in[13];
    const float* v2  = (const float*)d_in[14];
    float* out = (float*)d_out;

    // workspace layout (64-byte aligned sections)
    char* ws = (char*)d_ws;
    int*       idxbuf = (int*)ws;                                 // 4096*64*4 = 1,048,576
    _Float16*  W1h    = (_Float16*)(ws + 1048576);                // 256*288*2 =   147,456
    _Float16*  W2h    = (_Float16*)(ws + 1048576 + 147456);       // 256*256*2 =   131,072
    float*     bias1  = (float*)(ws + 1048576 + 147456 + 131072); //               1,024
    float*     bias2  = (float*)(ws + 1048576 + 147456 + 131072 + 1024);

    fold_weights_kernel<<<256, 256, 0, stream>>>(
        W1, b1, g1, be1, m1, v1, W2, b2, g2, be2, m2, v2,
        W1h, W2h, bias1, bias2);

    ball_query_kernel<<<(BB * MM) / 8, 256, 0, stream>>>(xyz, nxyz, idxbuf);

    fused_mlp_kernel<<<BB * MM, 256, 0, stream>>>(
        xyz, nxyz, feats, idxbuf, W1h, bias1, W2h, bias2, out);
}
